// GILEHead_73701638800241
// MI455X (gfx1250) — matrix operand order repaired and checked by reading, not compiled, not run on hardware
//
#include <hip/hip_runtime.h>

// CDNA5 / gfx1250, wave32. bf16 WMMA 16x16x32 with f32 accumulation.
// Pipeline:
//   0) one-shot f32->bf16 conversion of Wd, Wl into workspace
//   1) gemm_h64:  d  = relu(data  . Wd^T + bd)            -> bf16 [32768,64]
//                 l' = relu(label . Wl^T + bl) * Wo       -> bf16 [1024,64]
//   2) logits_h64: out = d @ l'^T + bo                    -> f32  [32768,1024]
// Inner loops are software-pipelined: next iteration's operands are prefetched
// before the current iteration's WMMAs, so loads get a full iteration of slack.

typedef __attribute__((ext_vector_type(4)))  float   v4f;
typedef __attribute__((ext_vector_type(8)))  float   v8f;
typedef __attribute__((ext_vector_type(8)))  __bf16  v8bf;
typedef __attribute__((ext_vector_type(16))) __bf16  v16bf;

#define WMMA_BF16(A, B, C) \
  __builtin_amdgcn_wmma_f32_16x16x32_bf16(false, (A), false, (B), (short)0, (C), false, false)

// Pack four f32x4 values into a v16bf fragment (element i = cast of i-th source value).
static __device__ inline v16bf cvt16(v4f x0, v4f x1, v4f x2, v4f x3) {
  v16bf r;
  r[0]  = (__bf16)x0[0]; r[1]  = (__bf16)x0[1]; r[2]  = (__bf16)x0[2]; r[3]  = (__bf16)x0[3];
  r[4]  = (__bf16)x1[0]; r[5]  = (__bf16)x1[1]; r[6]  = (__bf16)x1[2]; r[7]  = (__bf16)x1[3];
  r[8]  = (__bf16)x2[0]; r[9]  = (__bf16)x2[1]; r[10] = (__bf16)x2[2]; r[11] = (__bf16)x2[3];
  r[12] = (__bf16)x3[0]; r[13] = (__bf16)x3[1]; r[14] = (__bf16)x3[2]; r[15] = (__bf16)x3[3];
  return r;
}

// One-shot f32 -> bf16 conversion (8 elements per thread). n8 = n/8.
__global__ void __launch_bounds__(256)
cvt_f32_bf16(const float* __restrict__ src, __bf16* __restrict__ dst, int n8) {
  const int i = blockIdx.x * 256 + threadIdx.x;
  if (i < n8) {
    v4f x0 = *(const v4f*)(src + (size_t)i * 8);
    v4f x1 = *(const v4f*)(src + (size_t)i * 8 + 4);
    v8bf r;
    r[0] = (__bf16)x0[0]; r[1] = (__bf16)x0[1]; r[2] = (__bf16)x0[2]; r[3] = (__bf16)x0[3];
    r[4] = (__bf16)x1[0]; r[5] = (__bf16)x1[1]; r[6] = (__bf16)x1[2]; r[7] = (__bf16)x1[3];
    *(v8bf*)(dst + (size_t)i * 8) = r;
  }
}

// out[m, h] = relu( X[m,:768] . W[h,:768] + bias[h] ) * (hasScale ? scale[h] : 1)
// One wave -> 16 rows x all 64 hidden cols: A fragment (f32->bf16 inline, X read from
// HBM exactly once) reused across 4 pre-converted bf16 B fragments. 8 waves per block.
// A (16x32 bf16) lane layout: row = lane&15, K = (lane>>4)*16 + {0..15}.
// B (32x16 bf16) lane layout: col = lane&15, K = (lane>>4)*16 + {0..15}.
__global__ void __launch_bounds__(256)
gemm_h64(const float* __restrict__ X,        // [M, 768] row-major f32
         const __bf16* __restrict__ Wbf,     // [64, 768] row-major bf16 (pre-converted)
         const float* __restrict__ bias,     // [64]
         const float* __restrict__ scale,    // [64] or unused
         int hasScale,
         __bf16* __restrict__ out) {         // [M, 64] row-major bf16
  const int lane    = threadIdx.x & 31;
  const int wave    = threadIdx.x >> 5;               // 0..7
  const int rowTile = blockIdx.x * 8 + wave;          // 16 rows per wave
  const int half    = lane >> 4;                      // which K half-group
  const int mA      = rowTile * 16 + (lane & 15);     // A row for this lane
  const int kgA     = half * 16;                      // A K-subgroup base
  const int kgB     = half * 16;                      // B K-subgroup base
  const int hB      = lane & 15;                      // B column within 16-tile

  const float*  xrow  = X   + (size_t)mA * 768 + kgA;
  const __bf16* wbase = Wbf + (size_t)hB * 768 + kgB; // + t*16*768 per col tile

  v8f acc[4] = {};

  // ---- software pipeline: prologue loads for k0 = 0 ----
  v4f  ax0 = *(const v4f*)(xrow);
  v4f  ax1 = *(const v4f*)(xrow + 4);
  v4f  ax2 = *(const v4f*)(xrow + 8);
  v4f  ax3 = *(const v4f*)(xrow + 12);
  v16bf bc0 = *(const v16bf*)(wbase);
  v16bf bc1 = *(const v16bf*)(wbase + 12288);
  v16bf bc2 = *(const v16bf*)(wbase + 24576);
  v16bf bc3 = *(const v16bf*)(wbase + 36864);

  for (int k0 = 0; k0 < 768; k0 += 32) {
    // Prefetch next iteration's operands (wrap on last iter: valid addr, discarded).
    const int kn = (k0 + 32 < 768) ? (k0 + 32) : 0;
    v4f  nx0 = *(const v4f*)(xrow + kn);
    v4f  nx1 = *(const v4f*)(xrow + kn + 4);
    v4f  nx2 = *(const v4f*)(xrow + kn + 8);
    v4f  nx3 = *(const v4f*)(xrow + kn + 12);
    v16bf nb0 = *(const v16bf*)(wbase + kn);
    v16bf nb1 = *(const v16bf*)(wbase + 12288 + kn);
    v16bf nb2 = *(const v16bf*)(wbase + 24576 + kn);
    v16bf nb3 = *(const v16bf*)(wbase + 36864 + kn);

    // Compute on current (already-resident) operands.
    v16bf a = cvt16(ax0, ax1, ax2, ax3);
    acc[0] = WMMA_BF16(a, bc0, acc[0]);
    acc[1] = WMMA_BF16(a, bc1, acc[1]);
    acc[2] = WMMA_BF16(a, bc2, acc[2]);
    acc[3] = WMMA_BF16(a, bc3, acc[3]);

    ax0 = nx0; ax1 = nx1; ax2 = nx2; ax3 = nx3;
    bc0 = nb0; bc1 = nb1; bc2 = nb2; bc3 = nb3;
  }

#pragma unroll
  for (int t = 0; t < 4; ++t) {
    const int h  = t * 16 + hB;
    const float bi = bias[h];
    const float sc = hasScale ? scale[h] : 1.0f;
#pragma unroll
    for (int v = 0; v < 8; ++v) {
      const int m = rowTile * 16 + half * 8 + v;      // C/D row for (lane, v)
      float r = acc[t][v] + bi;
      r = fmaxf(r, 0.0f) * sc;
      out[(size_t)m * 64 + h] = (__bf16)r;
    }
  }
}

// logits[m, n] = Dm[m,:64] . Lm[n,:64] + bo   (K=64 -> two WMMAs per 16x16 tile)
// One block per 16-row tile; each of 8 waves handles 8 of the 64 label tiles.
// A fragments loaded once per wave; Lm (128 KB) and Dm (4 MB) are L2-resident.
__global__ void __launch_bounds__(256)
logits_h64(const __bf16* __restrict__ Dm,   // [32768, 64] bf16
           const __bf16* __restrict__ Lm,   // [1024, 64] bf16 (l * Wo folded)
           const float* __restrict__ bo,    // [1]
           float* __restrict__ out) {       // [32768, 1024] f32
  const int lane    = threadIdx.x & 31;
  const int wave    = threadIdx.x >> 5;
  const int rowTile = blockIdx.x;           // 0..2047
  const int half    = lane >> 4;
  const int mA      = rowTile * 16 + (lane & 15);
  const int kgA     = half * 16;
  const int kgB     = half * 16;
  const int nB      = lane & 15;

  const __bf16* drow = Dm + (size_t)mA * 64;
  v8bf a00 = *(const v8bf*)(drow + kgA);          // K = kgA+0..7
  v8bf a01 = *(const v8bf*)(drow + kgA + 8);      // K = kgA+8..15
  v8bf a10 = *(const v8bf*)(drow + 32 + kgA);     // K = 32+kgA+0..7
  v8bf a11 = *(const v8bf*)(drow + 32 + kgA + 8);
  v16bf a0 = __builtin_shufflevector(a00, a01, 0,1,2,3,4,5,6,7,8,9,10,11,12,13,14,15);
  v16bf a1 = __builtin_shufflevector(a10, a11, 0,1,2,3,4,5,6,7,8,9,10,11,12,13,14,15);

  const float bias = bo[0];

  // Wave handles nt = wave + it*8, it = 0..7. Double-buffer the B fragments.
  const __bf16* lptr = Lm + (size_t)(wave * 16 + nB) * 64 + kgB;
  const size_t  lstep = (size_t)8 * 16 * 64;      // 8 n-tiles forward
  v16bf b0 = *(const v16bf*)(lptr);
  v16bf b1 = *(const v16bf*)(lptr + 32);

  for (int it = 0; it < 8; ++it) {
    const __bf16* lnext = lptr + ((it < 7) ? lstep : 0);
    v16bf nb0 = *(const v16bf*)(lnext);
    v16bf nb1 = *(const v16bf*)(lnext + 32);

    v8f acc = {};
    acc = WMMA_BF16(a0, b0, acc);
    acc = WMMA_BF16(a1, b1, acc);

    const int ncol = (wave + it * 8) * 16 + nB;
#pragma unroll
    for (int v = 0; v < 8; ++v) {
      const int m = rowTile * 16 + half * 8 + v;
      out[(size_t)m * 1024 + ncol] = acc[v] + bias;
    }

    b0 = nb0; b1 = nb1;
    lptr = lnext;
  }
}

extern "C" void kernel_launch(void* const* d_in, const int* in_sizes, int n_in,
                              void* d_out, int out_size, void* d_ws, size_t ws_size,
                              hipStream_t stream) {
  const float* data = (const float*)d_in[0];  // [8,4096,768]
  const float* labf = (const float*)d_in[1];  // [1024,768]
  const float* Wd   = (const float*)d_in[2];  // [64,768]
  const float* bd   = (const float*)d_in[3];  // [64]
  const float* Wl   = (const float*)d_in[4];  // [64,768]
  const float* bl   = (const float*)d_in[5];  // [64]
  const float* Wo   = (const float*)d_in[6];  // [1,64]
  const float* bo   = (const float*)d_in[7];  // [1]
  float* out = (float*)d_out;                 // [8,4096,1024]

  // Workspace layout (bf16 elements)
  __bf16* dbf  = (__bf16*)d_ws;                       // 32768*64  = 4 MB
  __bf16* lbf  = dbf  + (size_t)32768 * 64;           // 1024*64   = 128 KB
  __bf16* wdbf = lbf  + (size_t)1024 * 64;            // 64*768    = 96 KB
  __bf16* wlbf = wdbf + (size_t)64 * 768;             // 64*768    = 96 KB

  // 0) Pre-convert weights to bf16 (49152 elems each -> 6144 threads -> 24 blocks)
  cvt_f32_bf16<<<24, 256, 0, stream>>>(Wd, wdbf, 6144);
  cvt_f32_bf16<<<24, 256, 0, stream>>>(Wl, wlbf, 6144);

  // 1) Label branch: l' = relu(label . Wl^T + bl) * Wo   (64 row tiles -> 8 blocks)
  gemm_h64<<<8, 256, 0, stream>>>(labf, wlbf, bl, Wo, 1, lbf);
  //    Data branch: d = relu(data . Wd^T + bd)           (2048 row tiles -> 256 blocks)
  gemm_h64<<<256, 256, 0, stream>>>(data, wdbf, bd, nullptr, 0, dbf);

  // 2) logits = d @ l'^T + bo                            (2048 row-tile blocks)
  logits_h64<<<2048, 256, 0, stream>>>(dbf, lbf, bo, out);
}
